// MLA_7258494730472
// MI455X (gfx1250) — matrix operand order repaired and checked by reading, not compiled, not run on hardware
//
#include <hip/hip_runtime.h>
#include <hip/hip_bf16.h>
#include <math.h>

// ---------------- problem constants (fixed by reference) ----------------
#define T_SEQ   2048
#define DMODEL  2048
#define NH      16
#define DHEAD   128
#define DROPE   64
#define DCONT   64
#define DLAT    512
#define SCALE_ATTN 0.08838834764831845f   // 1/sqrt(128)

typedef unsigned short u16;
typedef unsigned int   u32;
typedef unsigned long long u64;

typedef __attribute__((ext_vector_type(16))) __bf16 v16bf;
typedef __attribute__((ext_vector_type(16))) u16    v16u;
typedef __attribute__((ext_vector_type(8)))  float  v8f;

struct __align__(16) Q16 { u32 d[4]; };          // 16-byte move unit
union Frag32B { v16u v; Q16 q[2]; };             // one WMMA 16-half fragment
union Vec8H   { Q16 q; u16 u[8]; };              // 8 bf16 halves

__device__ __forceinline__ u16 f2bf(float f) {
  union { float f; u32 u; } c; c.f = f;
  u32 u = c.u;
  u32 r = u + 0x7FFFu + ((u >> 16) & 1u);        // round-to-nearest-even
  return (u16)(r >> 16);
}
__device__ __forceinline__ float bf2f(u16 h) {
  union { u32 u; float f; } c; c.u = ((u32)h) << 16; return c.f;
}
__device__ __forceinline__ v8f wmma_bf16(v16u a, v16u b, v8f c) {
  return __builtin_amdgcn_wmma_f32_16x16x32_bf16(
      false, __builtin_bit_cast(v16bf, a),
      false, __builtin_bit_cast(v16bf, b),
      (short)0, c, false, false);
}
// Async global->LDS 16B copy (GV mode), tracked with ASYNCcnt.
__device__ __forceinline__ void async_b128_to_lds(const void* lds, const void* g) {
  unsigned loff = (unsigned)(size_t)lds;          // LDS byte offset (as3 -> 32-bit)
  u64 ga = (u64)g;
  asm volatile("global_load_async_to_lds_b128 %0, %1, off"
               :: "v"(loff), "v"(ga) : "memory");
}
__device__ __forceinline__ void wait_async0() {
  asm volatile("s_wait_asynccnt 0x0" ::: "memory");
}

// ---------------- f32 -> bf16 convert ----------------
__global__ __launch_bounds__(256) void mla_f32_to_bf16(
    const float* __restrict__ in, u16* __restrict__ out, int n) {
  int i = blockIdx.x * 256 + threadIdx.x;
  if (i < n) out[i] = f2bf(in[i]);
}

// ---------------- bf16 WMMA GEMM: C[M,N] = A[M,K] @ B[K,N] ----------------
// Block tile 64x128, 8 waves, each wave one 16x64 tile (4 accumulators).
// Double-buffered LDS: async-load tile t+1 while WMMAs consume tile t,
// one s_wait_asynccnt + one barrier per K-step.
__global__ __launch_bounds__(256) void mla_gemm_bf16(
    const u16* __restrict__ A, const u16* __restrict__ B, void* __restrict__ C,
    int M, int N, int K, int f32out)
{
  __shared__ __align__(16) u16 As[2][64][32];      // [buf][m][k]
  __shared__ __align__(16) u16 Bst[2][128][32];    // [buf][n][k] (transposed)
  const int tid  = threadIdx.x;
  const int wid  = tid >> 5, lane = tid & 31;
  const int lrow = lane & 15, lhalf = lane >> 4;
  const int bm = blockIdx.y * 64, bn = blockIdx.x * 128;
  const int m0 = (wid & 3) * 16, n0 = (wid >> 2) * 64;
  const int ar = tid >> 2, ac = (tid & 3) * 8;     // A coop coords (64x4 groups)
  v8f acc[4] = {};

  auto stage = [&](int t, int b) {
    const int k0 = t << 5;
    // A tile: async global->LDS, row-major (no VGPR round-trip)
    async_b128_to_lds(&As[b][ar][ac], A + (size_t)(bm + ar) * K + k0 + ac);
    // B tile: load + transpose-scatter into [n][k]
#pragma unroll
    for (int it = 0; it < 2; ++it) {
      const int idx = tid + it * 256;
      const int r = idx >> 4, c = (idx & 15) * 8;
      Vec8H vb;
      vb.q = *(const Q16*)(B + (size_t)(k0 + r) * N + bn + c);
#pragma unroll
      for (int j = 0; j < 8; ++j) Bst[b][c + j][r] = vb.u[j];
    }
  };

  const int ntiles = K >> 5;
  stage(0, 0);
  wait_async0();
  __syncthreads();
  int buf = 0;
  for (int t = 0; t < ntiles; ++t) {
    if (t + 1 < ntiles) {
      stage(t + 1, buf ^ 1);
      if (t + 2 < ntiles)   // near-scope speculative prefetch of tile t+2
        __builtin_prefetch(A + (size_t)(bm + ar) * K + ((t + 2) << 5) + ac, 0, 3);
    }
    // A fragment: lane row = m0+lrow; K elems {kb..kb+15}, kb=16*lhalf
    Frag32B a;
    const int kb = lhalf ? 16 : 0;
    a.q[0] = *(const Q16*)&As[buf][m0 + lrow][kb];
    a.q[1] = *(const Q16*)&As[buf][m0 + lrow][kb + 8];
#pragma unroll
    for (int j = 0; j < 4; ++j) {
      Frag32B bfr;
      const u16* p = &Bst[buf][n0 + j * 16 + lrow][lhalf * 16];
      bfr.q[0] = *(const Q16*)p; bfr.q[1] = *(const Q16*)(p + 8);
      acc[j] = wmma_bf16(a.v, bfr.v, acc[j]);
    }
    wait_async0();
    __syncthreads();
    buf ^= 1;
  }

  // C layout: VGPR r, lanes 0-15 -> M=r, lanes 16-31 -> M=8+r; N = lrow
  const int row0 = bm + m0 + 8 * lhalf;
  const int c0   = bn + n0 + lrow;
#pragma unroll
  for (int r = 0; r < 8; ++r) {
    const size_t rr = (size_t)(row0 + r) * N;
    if (f32out) {
#pragma unroll
      for (int j = 0; j < 4; ++j) ((float*)C)[rr + c0 + j * 16] = acc[j][r];
    } else {
#pragma unroll
      for (int j = 0; j < 4; ++j) ((u16*)C)[rr + c0 + j * 16] = f2bf(acc[j][r]);
    }
  }
}

// ---------------- RoPE + pack q/k into (T, H*128) ----------------
__global__ __launch_bounds__(256) void mla_rope_pack(
    const u16* __restrict__ qc, const u16* __restrict__ qp,
    const u16* __restrict__ kc, const u16* __restrict__ kp,
    u16* __restrict__ qo, u16* __restrict__ ko)
{
  int idx = blockIdx.x * 256 + threadIdx.x;   // T*NH*32 pair-threads
  int i = idx & 31;
  int h = (idx >> 5) & (NH - 1);
  int t = idx >> 9;
  if (t >= T_SEQ) return;
  // inv_freq = theta^(-i/32); angle = t * inv_freq
  float inv = __expf(-(float)i * (9.2103403719761836f / 32.0f)); // ln(1e4)
  float ang = (float)t * inv;
  float sn, cs;
  __sincosf(ang, &sn, &cs);
  size_t src = (size_t)t * (NH * 64) + h * 64 + i;
  size_t dst = (size_t)t * (NH * DHEAD) + h * DHEAD;
  // content halves (raw copy)
  qo[dst + i]      = qc[src];
  qo[dst + 32 + i] = qc[src + 32];
  ko[dst + i]      = kc[src];
  ko[dst + 32 + i] = kc[src + 32];
  // rotated position halves
  float qx1 = bf2f(qp[src]), qx2 = bf2f(qp[src + 32]);
  qo[dst + 64 + i] = f2bf(qx1 * cs - qx2 * sn);
  qo[dst + 96 + i] = f2bf(qx1 * sn + qx2 * cs);
  float kx1 = bf2f(kp[src]), kx2 = bf2f(kp[src + 32]);
  ko[dst + 64 + i] = f2bf(kx1 * cs - kx2 * sn);
  ko[dst + 96 + i] = f2bf(kx1 * sn + kx2 * cs);
}

// ---------------- Flash attention: 4 waves x 16 q-rows, 1 head per block ----
// Double-buffered K/V staging (async K tile, transposed V tile).
__global__ __launch_bounds__(128) void mla_attention(
    const u16* __restrict__ Q, const u16* __restrict__ Kg,
    const u16* __restrict__ V, u16* __restrict__ O)
{
  __shared__ __align__(16) u16 Ks[2][32][128];     // [buf][kt][d]
  __shared__ __align__(16) u16 Vst[2][128][40];    // [buf][d][kt] (transposed, padded)
  __shared__ __align__(16) u16 Ps[4][16][32];      // per-wave P relayout scratch
  const int tid  = threadIdx.x;
  const int wid  = tid >> 5, lane = tid & 31;
  const int lrow = lane & 15, lhalf = lane >> 4;
  const int h    = blockIdx.y;
  const int qr0  = blockIdx.x * 64 + wid * 16;

  // preload Q fragments: 16 rows x 128 d = 4 WMMA K-chunks
  v16u qf[4];
  {
    const u16* qrow = Q + (size_t)(qr0 + lrow) * (NH * DHEAD) + h * DHEAD;
    const int kb = lhalf ? 16 : 0;
#pragma unroll
    for (int dc = 0; dc < 4; ++dc) {
      Frag32B a;
      a.q[0] = *(const Q16*)(qrow + dc * 32 + kb);
      a.q[1] = *(const Q16*)(qrow + dc * 32 + kb + 8);
      qf[dc] = a.v;
    }
  }
  v8f o[8] = {};                 // O tile 16x128 (8 N-subtiles)
  float mrow[8], srow[8];
#pragma unroll
  for (int r = 0; r < 8; ++r) { mrow[r] = -3.0e38f; srow[r] = 0.0f; }

  auto stage = [&](int t, int b) {
    const int kt0 = t << 5;
    // K tile: async global->LDS (row-major, ASYNCcnt path)
#pragma unroll
    for (int it = 0; it < 4; ++it) {
      const int idx = tid + it * 128;
      const int r = idx >> 4, c = (idx & 15) * 8;
      async_b128_to_lds(&Ks[b][r][c],
                        Kg + (size_t)(kt0 + r) * (NH * DHEAD) + h * DHEAD + c);
    }
    // V tile: load + transpose-scatter into [d][kt]
#pragma unroll
    for (int it = 0; it < 4; ++it) {
      const int idx = tid + it * 128;
      const int r = idx >> 4, c = (idx & 15) * 8;
      Vec8H vv;
      vv.q = *(const Q16*)(V + (size_t)(kt0 + r) * (NH * DHEAD) + h * DHEAD + c);
#pragma unroll
      for (int j = 0; j < 8; ++j) Vst[b][c + j][r] = vv.u[j];
    }
  };

  const int ntiles = (blockIdx.x + 1) * 2;     // causal: keys up to (qblock end)/32
  stage(0, 0);
  wait_async0();
  __syncthreads();
  int buf = 0;
  for (int t = 0; t < ntiles; ++t) {
    const int kt0 = t << 5;
    if (t + 1 < ntiles) stage(t + 1, buf ^ 1);

    // S = Q K^T : two 16x16 subtiles over 32 keys, 4 d-chunks each
    v8f s0 = {}, s1 = {};
#pragma unroll
    for (int dc = 0; dc < 4; ++dc) {
      Frag32B b0, b1;
      const u16* k0p = &Ks[buf][lrow][dc * 32 + lhalf * 16];
      b0.q[0] = *(const Q16*)k0p; b0.q[1] = *(const Q16*)(k0p + 8);
      const u16* k1p = &Ks[buf][16 + lrow][dc * 32 + lhalf * 16];
      b1.q[0] = *(const Q16*)k1p; b1.q[1] = *(const Q16*)(k1p + 8);
      s0 = wmma_bf16(qf[dc], b0.v, s0);
      s1 = wmma_bf16(qf[dc], b1.v, s1);
    }

    // online softmax (C layout: VGPR r -> row r + 8*lhalf, lane -> key lrow)
    float alpha[8];
#pragma unroll
    for (int r = 0; r < 8; ++r) {
      const int row  = r + 8 * lhalf;
      const int qabs = qr0 + row;
      float v0 = s0[r] * SCALE_ATTN;
      float v1 = s1[r] * SCALE_ATTN;
      const bool msk0 = (kt0 + lrow) > qabs;
      const bool msk1 = (kt0 + 16 + lrow) > qabs;
      v0 = msk0 ? -3.0e38f : v0;
      v1 = msk1 ? -3.0e38f : v1;
      float mx = fmaxf(v0, v1);
#pragma unroll
      for (int d = 1; d < 16; d <<= 1) mx = fmaxf(mx, __shfl_xor(mx, d, 32));
      const float mnew = fmaxf(mrow[r], mx);
      alpha[r] = __expf(mrow[r] - mnew);
      const float p0 = msk0 ? 0.0f : __expf(v0 - mnew);
      const float p1 = msk1 ? 0.0f : __expf(v1 - mnew);
      float ps = p0 + p1;
#pragma unroll
      for (int d = 1; d < 16; d <<= 1) ps += __shfl_xor(ps, d, 32);
      srow[r] = srow[r] * alpha[r] + ps;
      mrow[r] = mnew;
      Ps[wid][row][lrow]      = f2bf(p0);     // relayout via LDS
      Ps[wid][row][16 + lrow] = f2bf(p1);
    }
#pragma unroll
    for (int j = 0; j < 8; ++j)
#pragma unroll
      for (int r = 0; r < 8; ++r) o[j][r] *= alpha[r];

    // same-wave LDS RAW: wait on DS counter before fragment re-load
    asm volatile("s_wait_dscnt 0" ::: "memory");

    Frag32B pf;
    {
      const int kb = lhalf ? 16 : 0;
      pf.q[0] = *(const Q16*)&Ps[wid][lrow][kb];
      pf.q[1] = *(const Q16*)&Ps[wid][lrow][kb + 8];
    }
    // O += P V : 8 N-subtiles, K=32 keys (Vst transposed -> contiguous b128)
#pragma unroll
    for (int j = 0; j < 8; ++j) {
      Frag32B vf;
      const u16* vp = &Vst[buf][j * 16 + lrow][lhalf * 16];
      vf.q[0] = *(const Q16*)vp; vf.q[1] = *(const Q16*)(vp + 8);
      o[j] = wmma_bf16(pf.v, vf.v, o[j]);
    }

    wait_async0();
    __syncthreads();
    buf ^= 1;
  }

  // normalize and store bf16 attention output (T, H*128)
#pragma unroll
  for (int r = 0; r < 8; ++r) {
    const float inv = 1.0f / srow[r];
    u16* orow = O + (size_t)(qr0 + r + 8 * lhalf) * (NH * DHEAD) + h * DHEAD;
#pragma unroll
    for (int j = 0; j < 8; ++j) orow[j * 16 + lrow] = f2bf(o[j][r] * inv);
  }
}

// ---------------- host orchestration ----------------
extern "C" void kernel_launch(void* const* d_in, const int* in_sizes, int n_in,
                              void* d_out, int out_size, void* d_ws, size_t ws_size,
                              hipStream_t stream)
{
  (void)n_in; (void)out_size; (void)ws_size;
  const float* x    = (const float*)d_in[0];
  const float* Wqd  = (const float*)d_in[1];
  const float* Wqu  = (const float*)d_in[2];
  const float* Wqr  = (const float*)d_in[3];
  const float* Wkvd = (const float*)d_in[4];
  const float* Wku  = (const float*)d_in[5];
  const float* Wvu  = (const float*)d_in[6];
  const float* Wkr  = (const float*)d_in[7];
  const float* Wo   = (const float*)d_in[8];

  char* w = (char*)d_ws;
  auto bump = [&](size_t halves) {
    u16* p = (u16*)w;
    w += ((halves * 2 + 255) & ~(size_t)255);
    return p;
  };
  u16* xb    = bump((size_t)T_SEQ * DMODEL);
  u16* wqd   = bump((size_t)DMODEL * DLAT);
  u16* wqu   = bump((size_t)DLAT * NH * DCONT);
  u16* wqr   = bump((size_t)DMODEL * NH * DROPE);
  u16* wkvd  = bump((size_t)DMODEL * DLAT);
  u16* wku   = bump((size_t)DLAT * NH * DCONT);
  u16* wvu   = bump((size_t)DLAT * NH * DHEAD);
  u16* wkr   = bump((size_t)DMODEL * NH * DROPE);
  u16* wob   = bump((size_t)NH * DHEAD * DMODEL);
  u16* qlat  = bump((size_t)T_SEQ * DLAT);
  u16* kvlat = bump((size_t)T_SEQ * DLAT);
  u16* qc    = bump((size_t)T_SEQ * NH * DCONT);
  u16* qp    = bump((size_t)T_SEQ * NH * DROPE);
  u16* kc    = bump((size_t)T_SEQ * NH * DCONT);
  u16* kp    = bump((size_t)T_SEQ * NH * DROPE);
  u16* qbuf  = bump((size_t)T_SEQ * NH * DHEAD);
  u16* kbuf  = bump((size_t)T_SEQ * NH * DHEAD);
  u16* vbuf  = bump((size_t)T_SEQ * NH * DHEAD);
  u16* attn  = bump((size_t)T_SEQ * NH * DHEAD);

  auto cvt = [&](const float* src, u16* dst, int n) {
    mla_f32_to_bf16<<<(n + 255) / 256, 256, 0, stream>>>(src, dst, n);
  };
  cvt(x,    xb,   in_sizes[0]);
  cvt(Wqd,  wqd,  in_sizes[1]);
  cvt(Wqu,  wqu,  in_sizes[2]);
  cvt(Wqr,  wqr,  in_sizes[3]);
  cvt(Wkvd, wkvd, in_sizes[4]);
  cvt(Wku,  wku,  in_sizes[5]);
  cvt(Wvu,  wvu,  in_sizes[6]);
  cvt(Wkr,  wkr,  in_sizes[7]);
  cvt(Wo,   wob,  in_sizes[8]);

  auto gemm = [&](const u16* A, const u16* B, void* C, int M, int N, int K, int f32o) {
    mla_gemm_bf16<<<dim3(N / 128, M / 64), 256, 0, stream>>>(A, B, C, M, N, K, f32o);
  };
  gemm(xb,    wqd,  qlat, T_SEQ, DLAT,        DMODEL, 0);  // q latent
  gemm(qlat,  wqu,  qc,   T_SEQ, NH * DCONT,  DLAT,   0);  // q content
  gemm(xb,    wqr,  qp,   T_SEQ, NH * DROPE,  DMODEL, 0);  // q rope (pre-rot)
  gemm(xb,    wkvd, kvlat,T_SEQ, DLAT,        DMODEL, 0);  // kv latent
  gemm(kvlat, wku,  kc,   T_SEQ, NH * DCONT,  DLAT,   0);  // k content
  gemm(kvlat, wvu,  vbuf, T_SEQ, NH * DHEAD,  DLAT,   0);  // v (already packed)
  gemm(xb,    wkr,  kp,   T_SEQ, NH * DROPE,  DMODEL, 0);  // k rope (pre-rot)

  mla_rope_pack<<<(T_SEQ * NH * 32) / 256, 256, 0, stream>>>(qc, qp, kc, kp, qbuf, kbuf);

  mla_attention<<<dim3(T_SEQ / 64, NH), 128, 0, stream>>>(qbuf, kbuf, vbuf, attn);

  gemm(attn, wob, d_out, T_SEQ, DMODEL, NH * DHEAD, 1);    // final @ Wo, f32 out
}
